// PatchPartitioningModule_59554016526906
// MI455X (gfx1250) — matrix operand order repaired and checked by reading, not compiled, not run on hardware
//
#include <hip/hip_runtime.h>
#include <hip/hip_bf16.h>
#include <cstdint>
#include <cstddef>

typedef __attribute__((ext_vector_type(16))) _Float16 v16h;
typedef __attribute__((ext_vector_type(8)))  float    v8f;

// ---------------- problem constants ----------------
constexpr int BB = 4, CC = 3, HH = 256, WW = 256, PP = 64, NPTS = 64, KNB = 16;

// output region offsets (in floats), concatenated in reference return order
constexpr size_t NPATCH    = (size_t)BB * NPTS * CC * PP * PP;        // 3,145,728
constexpr size_t NCOORD    = (size_t)BB * NPTS * 2;                   // 512
constexpr size_t NGROUP    = (size_t)BB * NPTS * KNB * CC * PP * PP;  // 50,331,648
constexpr size_t NFLOW     = (size_t)BB * 2 * HH * WW;                // 524,288
constexpr size_t OFF_PATCH = 0;
constexpr size_t OFF_COORD = OFF_PATCH + NPATCH;
constexpr size_t OFF_GROUP = OFF_COORD + NCOORD;
constexpr size_t OFF_FLOW  = OFF_GROUP + NGROUP;
constexpr size_t OFF_DEF   = OFF_FLOW + NFLOW;
constexpr size_t OFF_BASE  = OFF_DEF + NFLOW;

// workspace layout (bytes)
constexpr size_t WS_A    = 0;                        // packed f16 A-fragments (<512KB)
constexpr size_t WS_BUF0 = 524288;                   // 8,388,608 floats
constexpr size_t WS_BUF1 = WS_BUF0 + 33554432;       // 8,388,608 floats
constexpr size_t WS_BUF2 = WS_BUF1 + 33554432;       // 2,097,152 floats
constexpr size_t WS_NBR  = WS_BUF2 + 8388608;        // 4096 ints

__device__ __forceinline__ int clampi(int v, int lo, int hi) {
    return v < lo ? lo : (v > hi ? hi : v);
}

// --------------------------------------------------------------------------
// Pack conv weights [Cout,CinReal,3,3] fp32 into f16 WMMA A-fragments with
// tap-major K ordering:  k = tap*Cinp + cin   (Cinp = pow2 padded Cin >= 32).
// A fragment order used here (16-bit A, 16x32 MxK; contiguous K, B uses the same):
//   lane l (m = l&15, h = l>>4): halves 0..15 -> K = h*16 + e
// --------------------------------------------------------------------------
__global__ void pack_w_kernel(const float* __restrict__ w, _Float16* __restrict__ dst,
                              int Cout, int CinReal, int Cinp, int log2Cinp,
                              int nmtiles, int nksteps) {
    int t = blockIdx.x * blockDim.x + threadIdx.x;
    int total = nmtiles * nksteps * 512;
    if (t >= total) return;
    int e     = t & 15;
    int lane  = (t >> 4) & 31;
    int kstep = (t >> 9) % nksteps;
    int mtile = t / (512 * nksteps);
    int m  = mtile * 16 + (lane & 15);
    int kk = kstep * 32 + (lane >> 4) * 16 + e;
    int tap = kk >> log2Cinp;              // 0..8 always (nksteps*32 == 9*Cinp)
    int cin = kk & (Cinp - 1);
    float v = 0.0f;
    if (m < Cout && cin < CinReal) {
        v = w[((m * CinReal + cin) * 3 + tap / 3) * 3 + (tap % 3)];
    }
    dst[t] = (_Float16)v;
}

// -------- zero-pad input channels: [B,3,H,W] -> [B,32,H,W] ----------------
__global__ void pad_x_kernel(const float* __restrict__ x, float* __restrict__ xp) {
    size_t total = (size_t)BB * 32 * HH * WW;
    size_t t = (size_t)blockIdx.x * blockDim.x + threadIdx.x;
    if (t >= total) return;
    size_t pix = t % ((size_t)HH * WW);
    int c = (int)((t / ((size_t)HH * WW)) % 32);
    int b = (int)(t / ((size_t)32 * HH * WW));
    xp[t] = (c < CC) ? x[((size_t)b * CC + c) * HH * WW + pix] : 0.0f;
}

// --------------------------------------------------------------------------
// Implicit-GEMM 3x3 SAME conv via V_WMMA_F32_16X16X32_F16.
// One wave computes a 16(Cout) x 32(pixels) tile: one A fragment, two B
// fragments, two WMMAs per K-step (independent -> back-to-back XDL issue).
// Tap-major K ordering makes tap/cin decode branch- and division-free:
//   tap     = kstep >> cpwShift            (constant over the 32-wide step)
//   cinbase = (kstep & cpwMask)*32 + khalf*16 ; cin = cinbase + e
// Zero padding handled by clamped address * {0,1} mask (per-step, not per-e).
// act: 0 none, 1 relu, 2 tanh
// --------------------------------------------------------------------------
__global__ void conv3x3_wmma_kernel(const float* __restrict__ in, float* __restrict__ out,
                                    const _Float16* __restrict__ packA,
                                    const float* __restrict__ bias,
                                    int Bn, int Cinp, int cpwShift, int cpwMask,
                                    int Cout, int H, int W,
                                    int nmtiles, int nksteps, int act) {
    const int lane = threadIdx.x & 31;
    const int gw   = (int)((blockIdx.x * blockDim.x + threadIdx.x) >> 5);
    const int nnt  = (H * W) >> 5;                  // 32-pixel groups
    const int totalWaves = Bn * nmtiles * nnt;
    if (gw >= totalWaves) return;                   // whole-wave exit

    const int ntile = gw % nnt;
    const int mtile = (gw / nnt) % nmtiles;
    const int bb    = gw / (nnt * nmtiles);
    const int nloc  = lane & 15;
    const int khalf = lane >> 4;
    const int p0    = ntile * 32 + nloc;            // 32 pixels, one row (W%32==0)
    const int oy  = p0 / W;
    const int ox0 = p0 % W;
    const int ox1 = ox0 + 16;
    const int HWs = H * W;
    const float* inB = in + (size_t)bb * Cinp * HWs;

    v8f acc0 = {}, acc1 = {};
    for (int kstep = 0; kstep < nksteps; ++kstep) {
        // A fragment: one aligned 32-byte vector load (pre-packed layout)
        v16h a = *(const v16h*)(packA + ((size_t)(mtile * nksteps + kstep) * 32 + lane) * 16);

        const int tap = kstep >> cpwShift;          // 0..8
        const int ky  = (tap >= 6) ? 2 : (tap >= 3 ? 1 : 0);
        const int kx  = tap - 3 * ky;
        const int cinbase = ((kstep & cpwMask) << 5) + (khalf << 4);

        const int iy  = oy + ky - 1;                // lane-uniform
        const bool rowok = (unsigned)iy < (unsigned)H;
        const int iyc = clampi(iy, 0, H - 1);

        const int ix0 = ox0 + kx - 1;
        const int ix1 = ox1 + kx - 1;
        const float mk0 = (rowok && (unsigned)ix0 < (unsigned)W) ? 1.0f : 0.0f;
        const float mk1 = (rowok && (unsigned)ix1 < (unsigned)W) ? 1.0f : 0.0f;
        const float* src0 = inB + ((size_t)cinbase * H + iyc) * W + clampi(ix0, 0, W - 1);
        const float* src1 = inB + ((size_t)cinbase * H + iyc) * W + clampi(ix1, 0, W - 1);

        v16h b0, b1;
        #pragma unroll
        for (int e = 0; e < 16; ++e) {
            b0[e] = (_Float16)(src0[(size_t)e * HWs] * mk0);
            b1[e] = (_Float16)(src1[(size_t)e * HWs] * mk1);
        }
        acc0 = __builtin_amdgcn_wmma_f32_16x16x32_f16(false, a, false, b0, (short)0, acc0, false, false);
        acc1 = __builtin_amdgcn_wmma_f32_16x16x32_f16(false, a, false, b1, (short)0, acc1, false, false);
    }

    // C/D layout: VGPR r, lanes 0-15 -> M=r, lanes 16-31 -> M=8+r; N = lane&15
    #pragma unroll
    for (int r = 0; r < 8; ++r) {
        int m = mtile * 16 + khalf * 8 + r;
        if (m < Cout) {
            float b = bias[m];
            float v0 = acc0[r] + b;
            float v1 = acc1[r] + b;
            if (act == 1)      { v0 = fmaxf(v0, 0.0f); v1 = fmaxf(v1, 0.0f); }
            else if (act == 2) { v0 = tanhf(v0);       v1 = tanhf(v1); }
            float* o = out + (((size_t)bb * Cout + m) * H + oy) * W;
            o[ox0] = v0;
            o[ox1] = v1;
        }
    }
}

// -------------------------- 2x2 max pool, stride 2 --------------------------
__global__ void maxpool2_kernel(const float* __restrict__ in, float* __restrict__ out,
                                int BC, int Hin, int Win) {
    int Ho = Hin >> 1, Wo = Win >> 1;
    size_t total = (size_t)BC * Ho * Wo;
    size_t t = (size_t)blockIdx.x * blockDim.x + threadIdx.x;
    if (t >= total) return;
    int x  = (int)(t % Wo);
    int y  = (int)((t / Wo) % Ho);
    int bc = (int)(t / ((size_t)Wo * Ho));
    const float* s = in + ((size_t)bc * Hin + 2 * y) * Win + 2 * x;
    out[t] = fmaxf(fmaxf(s[0], s[1]), fmaxf(s[Win], s[Win + 1]));
}

// ------------- 2x bilinear upsample (half-pixel, align_corners=False) -------------
__global__ void up2_kernel(const float* __restrict__ in, float* __restrict__ out,
                           int BC, int Hin, int Win) {
    int Ho = Hin * 2, Wo = Win * 2;
    size_t total = (size_t)BC * Ho * Wo;
    size_t t = (size_t)blockIdx.x * blockDim.x + threadIdx.x;
    if (t >= total) return;
    int x  = (int)(t % Wo);
    int y  = (int)((t / Wo) % Ho);
    int bc = (int)(t / ((size_t)Wo * Ho));
    float sy = (y + 0.5f) * 0.5f - 0.5f;
    float sx = (x + 0.5f) * 0.5f - 0.5f;
    int y0 = (int)floorf(sy); float wy = sy - (float)y0;
    int x0 = (int)floorf(sx); float wx = sx - (float)x0;
    int y0c = clampi(y0, 0, Hin - 1), y1c = clampi(y0 + 1, 0, Hin - 1);
    int x0c = clampi(x0, 0, Win - 1), x1c = clampi(x0 + 1, 0, Win - 1);
    const float* s = in + (size_t)bc * Hin * Win;
    float v = (1.f - wy) * ((1.f - wx) * s[(size_t)y0c * Win + x0c] + wx * s[(size_t)y0c * Win + x1c])
            +        wy  * ((1.f - wx) * s[(size_t)y1c * Win + x0c] + wx * s[(size_t)y1c * Win + x1c]);
    out[t] = v;
}

// ---------- base grid + deformed grid (reads flow region of d_out) ----------
__global__ void base_def_kernel(const float* __restrict__ temp, float* __restrict__ outbuf) {
    size_t total = (size_t)BB * HH * WW;
    size_t t = (size_t)blockIdx.x * blockDim.x + threadIdx.x;
    if (t >= total) return;
    int x = (int)(t % WW);
    int y = (int)((t / WW) % HH);
    int b = (int)(t / ((size_t)HH * WW));
    float gx = -1.0f + 2.0f * (float)x / (float)(WW - 1);
    float gy = -1.0f + 2.0f * (float)y / (float)(HH - 1);
    const float* flow = outbuf + OFF_FLOW;
    float f0 = flow[(((size_t)b * 2 + 0) * HH + y) * WW + x];
    float f1 = flow[(((size_t)b * 2 + 1) * HH + y) * WW + x];
    float tp = temp[0];
    size_t bi = (((size_t)b * HH + y) * WW + x) * 2;
    outbuf[OFF_BASE + bi]     = gx;
    outbuf[OFF_BASE + bi + 1] = gy;
    outbuf[OFF_DEF + bi]      = gx + f0 * tp;
    outbuf[OFF_DEF + bi + 1]  = gy + f1 * tp;
}

// -------- coords = deformed[:, ::32, ::32, :].reshape(B, 64, 2) --------
__global__ void coords_kernel(float* __restrict__ outbuf) {
    int t = blockIdx.x * blockDim.x + threadIdx.x;
    if (t >= BB * NPTS * 2) return;
    int c = t & 1;
    int n = (t >> 1) % NPTS;
    int b = t / (NPTS * 2);
    int yi = n / 8, xi = n % 8;
    outbuf[OFF_COORD + t] =
        outbuf[OFF_DEF + (((size_t)b * HH + yi * 32) * WW + xi * 32) * 2 + c];
}

// -------- top-K=16 nearest (squared dist, stable lowest-index ties) --------
__global__ void topk_kernel(const float* __restrict__ outbuf, int* __restrict__ nbr) {
    int t = blockIdx.x * blockDim.x + threadIdx.x;
    if (t >= BB * NPTS) return;
    int b = t / NPTS, n = t % NPTS;
    const float* cb = outbuf + OFF_COORD + (size_t)b * NPTS * 2;
    float cx = cb[n * 2], cy = cb[n * 2 + 1];
    float d2[NPTS];
    for (int j = 0; j < NPTS; ++j) {
        float dx = cx - cb[j * 2], dy = cy - cb[j * 2 + 1];
        d2[j] = dx * dx + dy * dy;
    }
    uint64_t used = 0;
    for (int k = 0; k < KNB; ++k) {
        int best = 0; float bd = 3.4e38f;
        for (int j = 0; j < NPTS; ++j) {
            if (!((used >> j) & 1ull) && d2[j] < bd) { bd = d2[j]; best = j; }
        }
        used |= 1ull << best;
        nbr[t * KNB + k] = best;
    }
}

// -------- bilinear patch extraction (grid_sample, zeros, align_corners=False) --------
__global__ void patches_kernel(const float* __restrict__ x, float* __restrict__ outbuf) {
    size_t t = (size_t)blockIdx.x * blockDim.x + threadIdx.x;
    if (t >= NPATCH) return;
    int px = (int)(t % PP);
    int py = (int)((t / PP) % PP);
    int c  = (int)((t / ((size_t)PP * PP)) % CC);
    int n  = (int)((t / ((size_t)PP * PP * CC)) % NPTS);
    int b  = (int)(t / ((size_t)PP * PP * CC * NPTS));
    const float* cb = outbuf + OFF_COORD + ((size_t)b * NPTS + n) * 2;
    float lx = -1.0f + 2.0f * (float)px / (float)(PP - 1);
    float ly = -1.0f + 2.0f * (float)py / (float)(PP - 1);
    float gx = cb[0] + lx * (2.0f / WW) * (PP * 0.5f);
    float gy = cb[1] + ly * (2.0f / HH) * (PP * 0.5f);
    float ix = ((gx + 1.0f) * WW - 1.0f) * 0.5f;
    float iy = ((gy + 1.0f) * HH - 1.0f) * 0.5f;
    float x0 = floorf(ix), y0 = floorf(iy);
    float wx1 = ix - x0, wx0 = 1.0f - wx1;
    float wy1 = iy - y0, wy0 = 1.0f - wy1;
    const float* img = x + ((size_t)b * CC + c) * HH * WW;
    auto samp = [&](float xc, float yc) -> float {
        if (xc < 0.0f || xc > (float)(WW - 1) || yc < 0.0f || yc > (float)(HH - 1)) return 0.0f;
        return img[(size_t)((int)yc) * WW + (int)xc];
    };
    float v = samp(x0, y0)             * wx0 * wy0
            + samp(x0 + 1.f, y0)       * wx1 * wy0
            + samp(x0, y0 + 1.f)       * wx0 * wy1
            + samp(x0 + 1.f, y0 + 1.f) * wx1 * wy1;
    outbuf[OFF_PATCH + t] = v;
}

// -------- groups[b,n,k] = patches[b, nbr[b,n,k]] : contiguous float4 copy --------
__global__ void groups_kernel(float* __restrict__ outbuf, const int* __restrict__ nbr) {
    constexpr size_t PATCH_F  = (size_t)CC * PP * PP;   // 12288 floats, contiguous
    constexpr size_t PATCH_V4 = PATCH_F / 4;            // 3072 float4
    size_t total = (size_t)BB * NPTS * KNB * PATCH_V4;
    size_t t = (size_t)blockIdx.x * blockDim.x + threadIdx.x;
    if (t >= total) return;
    size_t within = t % PATCH_V4;
    size_t g      = t / PATCH_V4;                       // (b*NPTS + n)*KNB + k
    int b = (int)(g / (KNB * NPTS));
    int idx = nbr[g];
    const float4* src = (const float4*)(outbuf + OFF_PATCH + ((size_t)b * NPTS + idx) * PATCH_F) + within;
    float4*       dst = (float4*)(outbuf + OFF_GROUP + g * PATCH_F) + within;
    *dst = *src;
}

// ===========================================================================
extern "C" void kernel_launch(void* const* d_in, const int* in_sizes, int n_in,
                              void* d_out, int out_size, void* d_ws, size_t ws_size,
                              hipStream_t stream) {
    const float* x = (const float*)d_in[0];
    const float* Wt[8]; const float* bi[8];
    for (int i = 0; i < 8; ++i) { Wt[i] = (const float*)d_in[1 + 2 * i]; bi[i] = (const float*)d_in[2 + 2 * i]; }
    const float* temp = (const float*)d_in[17];

    float* out = (float*)d_out;
    char*  ws  = (char*)d_ws;
    _Float16* Abase = (_Float16*)(ws + WS_A);
    float* buf0 = (float*)(ws + WS_BUF0);
    float* buf1 = (float*)(ws + WS_BUF1);
    float* buf2 = (float*)(ws + WS_BUF2);
    int*   nbr  = (int*)(ws + WS_NBR);

    // per-layer metadata (Cinp = pow2-padded input channels; tap-major K)
    const int    CinR[8]  = {3, 32, 32, 64, 64, 64, 32, 32};
    const int    Cinp_[8] = {32, 32, 32, 64, 64, 64, 32, 32};
    const int    l2Cp[8]  = {5, 5, 5, 6, 6, 6, 5, 5};
    const int    Cout_[8] = {32, 32, 64, 64, 64, 32, 32, 2};
    const int    Hl[8]    = {256, 256, 128, 128, 128, 128, 256, 256};
    const int    nmt[8]   = {2, 2, 4, 4, 4, 2, 2, 1};
    const int    nks[8]   = {9, 9, 9, 18, 18, 18, 9, 9};       // 9*Cinp/32
    const size_t aoff[8]  = {0, 9216, 18432, 36864, 73728, 110592, 129024, 138240};

    // 1) pack all conv weights into f16 WMMA A-fragments (tap-major K)
    for (int l = 0; l < 8; ++l) {
        int total = nmt[l] * nks[l] * 512;
        pack_w_kernel<<<(total + 255) / 256, 256, 0, stream>>>(
            Wt[l], Abase + aoff[l], Cout_[l], CinR[l], Cinp_[l], l2Cp[l], nmt[l], nks[l]);
    }

    auto conv = [&](int l, const float* in, float* o, int act) {
        int H = Hl[l];
        int waves = BB * nmt[l] * ((H * H) >> 5);   // always a multiple of 8
        int threads = waves * 32;
        conv3x3_wmma_kernel<<<threads / 256, 256, 0, stream>>>(
            in, o, Abase + aoff[l], bi[l], BB,
            Cinp_[l], l2Cp[l] - 5, (Cinp_[l] >> 5) - 1,
            Cout_[l], H, H, nmt[l], nks[l], act);
    };

    // 2) flow net (buf1 temporarily holds the 32-channel padded input)
    { size_t tot = (size_t)BB * 32 * HH * WW;
      pad_x_kernel<<<(unsigned)((tot + 255) / 256), 256, 0, stream>>>(x, buf1); }
    conv(0, buf1, buf0, 1);                                           // h1 [4,32,256,256]
    conv(1, buf0, buf1, 1);                                           // h2
    { size_t tot = (size_t)BB * 32 * 128 * 128;                        // p1 [4,32,128,128]
      maxpool2_kernel<<<(unsigned)((tot + 255) / 256), 256, 0, stream>>>(buf1, buf2, BB * 32, 256, 256); }
    conv(2, buf2, buf0, 1);                                           // h3 [4,64,128,128]
    conv(3, buf0, buf1, 1);                                           // h4
    { size_t tot = (size_t)BB * 64 * 64 * 64;                          // p2 [4,64,64,64]
      maxpool2_kernel<<<(unsigned)((tot + 255) / 256), 256, 0, stream>>>(buf1, buf2, BB * 64, 128, 128); }
    { size_t tot = (size_t)BB * 64 * 128 * 128;                        // u1 [4,64,128,128]
      up2_kernel<<<(unsigned)((tot + 255) / 256), 256, 0, stream>>>(buf2, buf0, BB * 64, 64, 64); }
    conv(4, buf0, buf1, 1);                                           // h5 [4,64,128,128]
    conv(5, buf1, buf0, 1);                                           // h6 [4,32,128,128]
    { size_t tot = (size_t)BB * 32 * 256 * 256;                        // u2 [4,32,256,256]
      up2_kernel<<<(unsigned)((tot + 255) / 256), 256, 0, stream>>>(buf0, buf1, BB * 32, 128, 128); }
    conv(6, buf1, buf0, 1);                                           // h7 [4,32,256,256]
    conv(7, buf0, out + OFF_FLOW, 2);                                 // flow = tanh(conv8)

    // 3) base grid + deformed grid
    { size_t tot = (size_t)BB * HH * WW;
      base_def_kernel<<<(unsigned)((tot + 255) / 256), 256, 0, stream>>>(temp, out); }

    // 4) coords
    coords_kernel<<<(BB * NPTS * 2 + 255) / 256, 256, 0, stream>>>(out);

    // 5) top-K neighbors
    topk_kernel<<<(BB * NPTS + 63) / 64, 64, 0, stream>>>(out, nbr);

    // 6) bilinear patch extraction
    { size_t tot = NPATCH;
      patches_kernel<<<(unsigned)((tot + 255) / 256), 256, 0, stream>>>(x, out); }

    // 7) neighbor-group gather (float4 streaming, L2-resident patch reads)
    { size_t tot = (size_t)BB * NPTS * KNB * ((size_t)CC * PP * PP / 4);
      groups_kernel<<<(unsigned)((tot + 255) / 256), 256, 0, stream>>>(out, nbr); }
}
